// S4AdvandedBlock_29712583754075
// MI455X (gfx1250) — matrix operand order repaired and checked by reading, not compiled, not run on hardware
//
#include <hip/hip_runtime.h>
#include <hip/hip_bf16.h>
#include <math.h>

#define BSZ   16
#define LEN   2048
#define DDIM  256
#define HDIM  64
#define NFREQ 1025          // L/2+1 frequency bins kept by irfft crop
#define KDIM  2112          // 66*32: [K_hat_re(1025) | K_hat_im(1..1023) | zero pad]
#define XSTRIDE 2056        // padded f16 row stride (4112 B = 16B-aligned, bank-spread)
#define PI_F 3.14159265358979323846f

typedef _Float16 h8   __attribute__((ext_vector_type(8)));
typedef _Float16 v16h __attribute__((ext_vector_type(16)));
typedef float    v8f  __attribute__((ext_vector_type(8)));

// ---- fragment loaders -------------------------------------------------------
// A-matrix 16x32 f16 (ISA 7.12.2): per lane, elements 0..7 = K {base..base+7},
// elements 8..15 = K {base+16..base+23}; base = 0 (lanes 0-15) or 8 (lanes 16-31).
__device__ __forceinline__ v16h ld16_split(const _Float16* p) {
  h8 c0 = *(const h8*)p;
  h8 c1 = *(const h8*)(p + 16);
  return __builtin_shufflevector(c0, c1, 0,1,2,3,4,5,6,7,8,9,10,11,12,13,14,15);
}
// B 32x16 f16, contiguous k order chosen here (A matches it): 16 K per lane (0..15 low half-wave,
// 16..31 high half-wave), column = lane&15.
__device__ __forceinline__ v16h ld16_contig(const _Float16* p) {
  h8 c0 = *(const h8*)p;
  h8 c1 = *(const h8*)(p + 8);
  return __builtin_shufflevector(c0, c1, 0,1,2,3,4,5,6,7,8,9,10,11,12,13,14,15);
}

// ---- Kernel 1: DPLR Cauchy kernel -> K_hat (D x 1025 complex) --------------
__global__ __launch_bounds__(256)
void cauchy_khat_kernel(const float* __restrict__ B, const float* __restrict__ Ct,
                        const float* __restrict__ Lre, const float* __restrict__ Lim,
                        const float* __restrict__ Pre, const float* __restrict__ Pim,
                        const float* __restrict__ Qre, const float* __restrict__ Qim,
                        float* __restrict__ khr, float* __restrict__ khi) {
  __shared__ float w00[HDIM], w01r[HDIM], w01i[HDIM], w10r[HDIM], w10i[HDIM],
                   w11r[HDIM], w11i[HDIM], lr[HDIM], li[HDIM];
  const int d = blockIdx.x;
  const int tid = threadIdx.x;
  if (tid < HDIM) {
    const int idx = d * HDIM + tid;
    const float b  = B[idx],  ct = Ct[idx];
    const float pr = Pre[idx], pi = Pim[idx];
    const float qr = Qre[idx], qi = Qim[idx];
    w00[tid]  = ct * b;                                 // C~ * B (real)
    w01r[tid] = ct * pr;  w01i[tid] = ct * pi;          // C~ * P
    w10r[tid] = qr * b;   w10i[tid] = -qi * b;          // conj(Q) * B
    w11r[tid] = qr * pr + qi * pi;                      // conj(Q) * P
    w11i[tid] = qr * pi - qi * pr;
    lr[tid] = Lre[idx];   li[tid] = Lim[idx];
  }
  __syncthreads();
  for (int l = tid; l < NFREQ; l += 256) {
    const float ang = 2.f * PI_F * (float)l / (float)LEN;
    const float wr = cosf(ang), wi = sinf(ang);
    const float apr = 1.f + wr, api = wi;     // 1 + omega
    const float amr = 1.f - wr, ami = -wi;    // 1 - omega
    const float inv = 1.f / (apr * apr + api * api);
    const float gr  = 20.f * (amr * apr + ami * api) * inv;   // (2/DT)(1-w)/(1+w)
    const float gi  = 20.f * (ami * apr - amr * api) * inv;
    const float cre = 2.f * apr * inv, cim = -2.f * api * inv; // 2/(1+omega)
    float k00r=0.f,k00i=0.f,k01r=0.f,k01i=0.f,k10r=0.f,k10i=0.f,k11r=0.f,k11i=0.f;
#pragma unroll 4
    for (int h = 0; h < HDIM; ++h) {
      const float dr = gr - lr[h], di = gi - li[h];
      const float r2 = 1.f / (dr * dr + di * di);
      const float rr = dr * r2, ri = -di * r2;   // 1/(g - Lambda)
      k00r += w00[h]*rr;                 k00i += w00[h]*ri;
      k01r += w01r[h]*rr - w01i[h]*ri;   k01i += w01r[h]*ri + w01i[h]*rr;
      k10r += w10r[h]*rr - w10i[h]*ri;   k10i += w10r[h]*ri + w10i[h]*rr;
      k11r += w11r[h]*rr - w11i[h]*ri;   k11i += w11r[h]*ri + w11i[h]*rr;
    }
    const float t1r = 1.f + k11r, t1i = k11i;
    const float t2r = k01r*t1r - k01i*t1i, t2i = k01r*t1i + k01i*t1r;
    const float t3r = t2r*k10r - t2i*k10i, t3i = t2r*k10i + t2i*k10r;
    const float kr = k00r - t3r, ki = k00i - t3i;
    khr[d * NFREQ + l] = cre * kr - cim * ki;
    khi[d * NFREQ + l] = cre * ki + cim * kr;
  }
}

// ---- Kernel 2a: build transposed irfft twiddle matrix Bt[t][l] (f16) -------
// K[t] = (1/L)(Xr0 + (-1)^t Xr_{L/2} + 2*sum_{l=1}^{L/2-1}(Xr_l cos - Xi_l sin))
__global__ __launch_bounds__(256)
void build_dft_b_kernel(_Float16* __restrict__ Bt) {
  const int t = blockIdx.x;
  for (int l = threadIdx.x; l < KDIM; l += 256) {
    float v = 0.f;
    if (l <= 1024) {
      const float w = (l == 0 || l == 1024) ? 1.f : 2.f;
      v = w * cosf(2.f * PI_F * (float)((l * t) & (LEN - 1)) / (float)LEN)
            * (1.f / (float)LEN);
    } else if (l < 2048) {
      const int lp = l - 1024;                 // 1..1023
      v = -2.f * sinf(2.f * PI_F * (float)((lp * t) & (LEN - 1)) / (float)LEN)
               * (1.f / (float)LEN);
    }
    Bt[(size_t)t * KDIM + l] = (_Float16)v;
  }
}

// ---- Kernel 2b: pack A = [K_hat_re | K_hat_im | 0] (D x KDIM, f16) ---------
__global__ __launch_bounds__(256)
void pack_a_kernel(const float* __restrict__ khr, const float* __restrict__ khi,
                   _Float16* __restrict__ A) {
  const int d = blockIdx.x;
  for (int c = threadIdx.x; c < KDIM; c += 256) {
    float v = 0.f;
    if (c <= 1024)      v = khr[d * NFREQ + c];
    else if (c < 2048)  v = khi[d * NFREQ + (c - 1024)];
    A[(size_t)d * KDIM + c] = (_Float16)v;
  }
}

// ---- Kernel 2c: irfft as WMMA GEMM: Kf(256x2048) = A(256x2112) * B(2112x2048)
__global__ __launch_bounds__(256)
void dft_gemm_kernel(const _Float16* __restrict__ A, const _Float16* __restrict__ Bt,
                     _Float16* __restrict__ Kf) {
  const int lane  = threadIdx.x & 31;
  const int wave  = threadIdx.x >> 5;
  const int mtile = blockIdx.x;                 // 0..15  (d tiles)
  const int ntile = blockIdx.y * 8 + wave;      // 0..127 (t tiles)
  const int m  = lane & 15;
  const int hi = lane >> 4;
  const int kA = hi * 16;
  const int kB = hi * 16;
  const _Float16* arow = A  + (size_t)(mtile * 16 + m) * KDIM;
  const _Float16* brow = Bt + (size_t)(ntile * 16 + m) * KDIM;  // column n = lane&15
  v8f acc = {};
  for (int c = 0; c < KDIM / 32; ++c) {
    v16h a = ld16_contig(arow + c * 32 + kA);
    v16h b = ld16_contig(brow + c * 32 + kB);
    acc = __builtin_amdgcn_wmma_f32_16x16x32_f16(false, a, false, b,
                                                 (short)0, acc, false, false);
  }
#pragma unroll
  for (int r = 0; r < 8; ++r) {
    const int dout = mtile * 16 + r + hi * 8;
    const int tout = ntile * 16 + m;
    Kf[(size_t)dout * LEN + tout] = (_Float16)acc[r];
  }
}

// ---- Kernel 3: causal convolution as blocked Toeplitz WMMA GEMM ------------
// Two blocks per d (split by output-tile parity). Each block:
//   1) stages x[:, :, d] as f16 in LDS (padded stride; L2 serves the gather),
//   2) builds the 64 parity-matching Toeplitz B-fragments in LDS, already in
//      WMMA lane order, so the hot loop reads B via two aligned ds_load_b128
//      (coalesced, bank-conflict-free) instead of 16 scalar u16 loads,
//   3) processes 4 output tiles per wave sharing each A-fragment load, only
//      over the causal chunks c <= (16j+15)/32.
__global__ __launch_bounds__(256)
void conv_wmma_kernel(const float* __restrict__ x, const _Float16* __restrict__ Kf,
                      float* __restrict__ y) {
  __shared__ alignas(16) _Float16 xs[BSZ * XSTRIDE];   // 65792 B
  __shared__ alignas(16) _Float16 fragB[64 * 512];     // 65536 B (64 Toeplitz tiles)
  const int parity = blockIdx.x & 1;
  const int d      = blockIdx.x >> 1;
  const int tid    = threadIdx.x;

  // stage x -> LDS (f16)
  for (int i = tid; i < BSZ * LEN; i += 256) {
    const int b = i >> 11, l = i & (LEN - 1);
    xs[b * XSTRIDE + l] = (_Float16)x[((size_t)b * LEN + l) * DDIM + d];
  }
  // build Toeplitz fragments: tile dd (= j - 2c, parity fixed), element
  // (k = kb+e, n): value = K[16*dd + n - kb - e] (0 outside [0, L)).
  // Stored per-lane-contiguous: fragB[dd2*512 + lane*16 + e].
  const _Float16* krow = Kf + (size_t)d * LEN;
  for (int i = tid; i < 64 * 512; i += 256) {
    const int dd2 = i >> 9;                // 0..63
    const int dd  = 2 * dd2 + parity;      // 0..127, parity class
    const int r   = i & 511;
    const int ln  = r >> 4;                // lane 0..31
    const int e   = r & 15;
    const int n   = ln & 15;
    const int kb  = (ln >> 4) * 16;
    const int kidx = 16 * dd + n - kb - e; // in [-31, 2047]
    fragB[i] = (kidx >= 0 && kidx < LEN) ? krow[kidx] : (_Float16)0.f;
  }
  __syncthreads();

  const int lane = tid & 31;
  const int wave = tid >> 5;
  const int m  = lane & 15;
  const int hi = lane >> 4;
  const int kA = hi * 16;
  const _Float16* xrow = &xs[m * XSTRIDE];
  const int lane16 = lane * 16;

  // wave handles j = parity + 2*wave + 16*k, k = 0..7, in two groups of 4
  for (int g = 0; g < 2; ++g) {
    const int k0 = g * 4;
    const int j0 = parity + 2 * wave + 16 * k0;     // group's 4 tiles: j0 + {0,16,32,48}
    const int b0 = wave + 8 * k0;                   // fragB base dd2 = b0 + 8*i - c
    const int cm0 = (16 *  j0       + 15) >> 5;
    const int cm1 = (16 * (j0 + 16) + 15) >> 5;
    const int cm2 = (16 * (j0 + 32) + 15) >> 5;
    const int cm3 = (16 * (j0 + 48) + 15) >> 5;
    v8f acc0 = {}, acc1 = {}, acc2 = {}, acc3 = {};
    for (int c = 0; c <= cm3; ++c) {
      const v16h a = ld16_contig(xrow + c * 32 + kA);   // shared across 4 tiles
      if (c <= cm0) {
        v16h b = ld16_contig(&fragB[(size_t)(b0      - c) * 512 + lane16]);
        acc0 = __builtin_amdgcn_wmma_f32_16x16x32_f16(false, a, false, b,
                                                      (short)0, acc0, false, false);
      }
      if (c <= cm1) {
        v16h b = ld16_contig(&fragB[(size_t)(b0 +  8 - c) * 512 + lane16]);
        acc1 = __builtin_amdgcn_wmma_f32_16x16x32_f16(false, a, false, b,
                                                      (short)0, acc1, false, false);
      }
      if (c <= cm2) {
        v16h b = ld16_contig(&fragB[(size_t)(b0 + 16 - c) * 512 + lane16]);
        acc2 = __builtin_amdgcn_wmma_f32_16x16x32_f16(false, a, false, b,
                                                      (short)0, acc2, false, false);
      }
      {
        v16h b = ld16_contig(&fragB[(size_t)(b0 + 24 - c) * 512 + lane16]);
        acc3 = __builtin_amdgcn_wmma_f32_16x16x32_f16(false, a, false, b,
                                                      (short)0, acc3, false, false);
      }
    }
#pragma unroll
    for (int r = 0; r < 8; ++r) {
      const int bo = r + hi * 8;
      y[((size_t)bo * LEN + (j0      ) * 16 + m) * DDIM + d] = acc0[r];
      y[((size_t)bo * LEN + (j0 + 16) * 16 + m) * DDIM + d] = acc1[r];
      y[((size_t)bo * LEN + (j0 + 32) * 16 + m) * DDIM + d] = acc2[r];
      y[((size_t)bo * LEN + (j0 + 48) * 16 + m) * DDIM + d] = acc3[r];
    }
  }
}

// ---- host ------------------------------------------------------------------
extern "C" void kernel_launch(void* const* d_in, const int* in_sizes, int n_in,
                              void* d_out, int out_size, void* d_ws, size_t ws_size,
                              hipStream_t stream) {
  const float* x   = (const float*)d_in[0];
  const float* B   = (const float*)d_in[1];
  const float* Ct  = (const float*)d_in[2];
  const float* Lre = (const float*)d_in[3];
  const float* Lim = (const float*)d_in[4];
  const float* Pre = (const float*)d_in[5];
  const float* Pim = (const float*)d_in[6];
  const float* Qre = (const float*)d_in[7];
  const float* Qim = (const float*)d_in[8];
  float* y = (float*)d_out;

  char* ws = (char*)d_ws;
  size_t off = 0;
  float* khr = (float*)(ws + off); off += (size_t)DDIM * NFREQ * sizeof(float);
  float* khi = (float*)(ws + off); off += (size_t)DDIM * NFREQ * sizeof(float);
  off = (off + 255) & ~(size_t)255;
  _Float16* Af = (_Float16*)(ws + off); off += (size_t)DDIM * KDIM * sizeof(_Float16);
  _Float16* Bt = (_Float16*)(ws + off); off += (size_t)LEN  * KDIM * sizeof(_Float16);
  _Float16* Kf = (_Float16*)(ws + off); off += (size_t)DDIM * LEN  * sizeof(_Float16);
  (void)ws_size; (void)in_sizes; (void)n_in; (void)out_size;  // ~12.9 MB scratch used

  cauchy_khat_kernel<<<DDIM, 256, 0, stream>>>(B, Ct, Lre, Lim, Pre, Pim, Qre, Qim,
                                               khr, khi);
  build_dft_b_kernel<<<LEN, 256, 0, stream>>>(Bt);
  pack_a_kernel<<<DDIM, 256, 0, stream>>>(khr, khi, Af);
  dim3 g2(16, 16);
  dft_gemm_kernel<<<g2, 256, 0, stream>>>(Af, Bt, Kf);
  conv_wmma_kernel<<<DDIM * 2, 256, 0, stream>>>(x, Kf, y);
}
